// WindowAttentionGlobal_14697378087071
// MI455X (gfx1250) — matrix operand order repaired and checked by reading, not compiled, not run on hardware
//
#include <hip/hip_runtime.h>
#include <hip/hip_bf16.h>

typedef _Float16 f16;
typedef __attribute__((ext_vector_type(16))) _Float16 v16h;
typedef __attribute__((ext_vector_type(8)))  _Float16 h8;
typedef __attribute__((ext_vector_type(8)))  float    v8f;

#define WMMA16(a,b,c) __builtin_amdgcn_wmma_f32_16x16x32_f16(false,(a),false,(b),(short)0,(c),false,false)

// ---------------- constants ----------------
// b=8 batch, m=2 agents, gx=gy=8 windows, w=7, d=256, h=8 heads, hd=32, N=98
// stage1 spatial 56, stage2 28, pooled 14. NW = 8*64 = 512 windows.

__device__ __forceinline__ v8f v8f_zero(){ v8f z;
#pragma unroll
  for(int i=0;i<8;++i) z[i]=0.0f; return z; }
__device__ __forceinline__ h8 h8_zero(){ h8 z;
#pragma unroll
  for(int i=0;i<8;++i) z[i]=(f16)0.0f; return z; }

// ---- CDNA5 async global->LDS DMA (tracked by ASYNCcnt) ----
__device__ __forceinline__ unsigned lds_off(const void* p){
  return (unsigned)(unsigned long long)p;   // shared aperture: low 32 bits = LDS offset
}
__device__ __forceinline__ void async_b128(unsigned lds, const void* g){
  asm volatile("global_load_async_to_lds_b128 %0, %1, off" :: "v"(lds), "v"(g) : "memory");
}
__device__ __forceinline__ void wait_async0(){
  asm volatile("s_wait_asynccnt 0" ::: "memory");
}

// A fragment: 16x32 f16, row-major LDS panel, stride ld halves (mult of 8)
__device__ __forceinline__ v16h frag_a(const f16* base,int ld,int m0,int k0,int lane){
  int row=m0+(lane&15), hi=lane>>4;
  const f16* p=base+row*ld+k0+hi*8;
  h8 x0=*(const h8*)p;
  h8 x1=*(const h8*)(p+16);
  v16h r;
#pragma unroll
  for(int t=0;t<8;++t){ r[t]=x0[t]; r[t+8]=x1[t]; }
  return r;
}
// B fragment: 32x16 f16 from B^T panel [n][k], stride ld halves
__device__ __forceinline__ v16h frag_b(const f16* base,int ld,int n0,int k0,int lane){
  int col=n0+(lane&15), hi=lane>>4;
  const f16* p=base+col*ld+k0+hi*8;
  h8 x0=*(const h8*)p;
  h8 x1=*(const h8*)(p+16);
  v16h r;
#pragma unroll
  for(int t=0;t<8;++t){ r[t]=x0[t]; r[t+8]=x1[t]; }
  return r;
}
__device__ __forceinline__ float gelu_f(float x){ return 0.5f*x*(1.0f+erff(x*0.70710678118654752f)); }

// ---------------- prep: weight transpose/convert + zero q_global ----------------
__global__ void k_prep(const float* __restrict__ qkvw, const float* __restrict__ tow,
                       const float* __restrict__ pw1, const float* __restrict__ pw2,
                       f16* __restrict__ qkvT, f16* __restrict__ toT,
                       f16* __restrict__ pw1h, f16* __restrict__ pw2h,
                       float* __restrict__ qg){
  int id = blockIdx.x*256 + threadIdx.x;
  if (id < 512*256){ int o=id>>8, i=id&255; qkvT[id]=(f16)qkvw[i*512+o]; }
  if (id < 256*256){
    int o=id>>8, i=id&255;
    toT[id]=(f16)tow[i*256+o];
    pw1h[id]=(f16)pw1[id];   // pw layout already [co][ci]
    pw2h[id]=(f16)pw2[id];
  }
  if (id < 8*98*256) qg[id]=0.0f;
}

// ---------------- rearrange input -> (16,256,56,56) ----------------
__global__ void k_rearrange_in(const float* __restrict__ x, float* __restrict__ xg0){
  long idx = (long)blockIdx.x*256 + threadIdx.x;
  const long total = 16L*256*56*56;
  if (idx >= total) return;
  int Y = (int)(idx % 56); long t = idx/56;
  int X = (int)(t % 56); t/=56;
  int c = (int)(t & 255); int bm = (int)(t>>8);
  int gx=X/7, w1=X%7, gy=Y/7, w2=Y%7;
  long src = ((((long)bm*8+gx)*8+gy)*49 + w1*7+w2)*256 + c;
  xg0[idx] = x[src];
}

// ---------------- depthwise 3x3 + gelu ----------------
__global__ void k_dwconv_gelu(const float* __restrict__ xin, const float* __restrict__ dw,
                              float* __restrict__ y, int Hs){
  long total = 16L*256*Hs*Hs;
  long idx = (long)blockIdx.x*256 + threadIdx.x;
  if (idx >= total) return;
  int Y=(int)(idx%Hs); long t=idx/Hs;
  int X=(int)(t%Hs); t/=Hs;
  int c=(int)(t&255); int bm=(int)(t>>8);
  const float* wp = dw + c*9;
  const float* ip = xin + ((long)bm*256 + c)*Hs*Hs;
  float acc=0.0f;
#pragma unroll
  for(int i=0;i<3;++i){
    int xx=X-1+i; if(xx<0||xx>=Hs) continue;
#pragma unroll
    for(int j=0;j<3;++j){
      int yy=Y-1+j; if(yy<0||yy>=Hs) continue;
      acc += wp[i*3+j]*ip[(long)xx*Hs+yy];
    }
  }
  y[idx] = gelu_f(acc);
}

// ---------------- channel mean ----------------
__global__ void k_chan_mean(const float* __restrict__ y, float* __restrict__ s0, int HW){
  __shared__ float red[256];
  int bc = blockIdx.x;
  const float* p = y + (long)bc*HW;
  float s=0.0f;
  for(int i=threadIdx.x;i<HW;i+=256) s+=p[i];
  red[threadIdx.x]=s; __syncthreads();
  for(int off=128;off>0;off>>=1){ if(threadIdx.x<off) red[threadIdx.x]+=red[threadIdx.x+off]; __syncthreads(); }
  if(threadIdx.x==0) s0[bc]=red[0]/(float)HW;
}

// ---------------- SE MLP ----------------
__global__ void k_se(const float* __restrict__ s0, const float* __restrict__ se1,
                     const float* __restrict__ se2, float* __restrict__ sout){
  __shared__ float sr[256];
  __shared__ float hid[64];
  int bm=blockIdx.x, t=threadIdx.x;
  sr[t]=s0[bm*256+t];
  __syncthreads();
  if(t<64){ float a=0.0f; for(int i=0;i<256;++i) a+=sr[i]*se1[i*64+t]; hid[t]=gelu_f(a); }
  __syncthreads();
  float a=0.0f;
  for(int j=0;j<64;++j) a+=hid[j]*se2[j*256+t];
  sout[bm*256+t]=1.0f/(1.0f+expf(-a));
}

// ---------------- transpose + SE-scale + f16 convert: yt[bm][p][c] ----------------
__global__ void k_yscale_t(const float* __restrict__ y, const float* __restrict__ s,
                           f16* __restrict__ yt, int HW){
  long total = 16L*HW*256;
  long idx = (long)blockIdx.x*256 + threadIdx.x;
  if(idx>=total) return;
  int c=(int)(idx&255); long t=idx>>8;
  int p=(int)(t%HW); int bm=(int)(t/HW);
  yt[idx]=(f16)(y[((long)bm*256+c)*HW+p]*s[bm*256+c]);
}

// ---------------- pointwise conv GEMM + residual (in-place) ----------------
__global__ __launch_bounds__(256) void k_pwconv(const f16* __restrict__ yt, const f16* __restrict__ pwh,
                                                float* __restrict__ xio, int P){
  __shared__ __align__(16) f16 As[64][40];
  __shared__ __align__(16) f16 Bs[64][40];
  int mbase = blockIdx.x*64, nbase = blockIdx.y*64, bm = blockIdx.z;
  int tid=threadIdx.x, lane=tid&31, wv=tid>>5;
  int wm=wv>>1, wn=wv&1;
  v8f acc0=v8f_zero(), acc1=v8f_zero();
  const f16* Ab = yt + (long)bm*P*256;
  for(int k0=0;k0<256;k0+=32){
    int row=tid>>2, ch=tid&3;
    int gr=mbase+row;
    if(gr<P) async_b128(lds_off(&As[row][ch*8]), Ab+(long)gr*256+k0+ch*8);
    else     *(h8*)&As[row][ch*8]=h8_zero();
    async_b128(lds_off(&Bs[row][ch*8]), pwh+(long)(nbase+row)*256+k0+ch*8);
    wait_async0();
    __syncthreads();
    v16h a  = frag_a(&As[0][0],40,wm*16,0,lane);
    v16h b0 = frag_b(&Bs[0][0],40,wn*32,0,lane);
    v16h b1 = frag_b(&Bs[0][0],40,wn*32+16,0,lane);
    acc0=WMMA16(a,b0,acc0); acc1=WMMA16(a,b1,acc1);
    __syncthreads();
  }
  int hi=lane>>4, nn=lane&15;
#pragma unroll
  for(int r=0;r<8;++r){
    int p = mbase + wm*16 + r + 8*hi;
    if(p<P){
      int c0=nbase+wn*32+nn;
      long o0=((long)bm*256+c0)*P+p;
      long o1=((long)bm*256+c0+16)*P+p;
      xio[o0]+=acc0[r];
      xio[o1]+=acc1[r];
    }
  }
}

// ---------------- maxpool 3x3 s2 p1 ----------------
__global__ void k_maxpool(const float* __restrict__ xin, float* __restrict__ xout, int Hi){
  int Ho=Hi/2;
  long total=16L*256*Ho*Ho;
  long idx=(long)blockIdx.x*256+threadIdx.x;
  if(idx>=total) return;
  int wo=(int)(idx%Ho); long t=idx/Ho;
  int ho=(int)(t%Ho); long bc=t/Ho;
  const float* p=xin+(long)bc*Hi*Hi;
  float m=-__builtin_inff();
#pragma unroll
  for(int di=0;di<3;++di){
    int hh=2*ho-1+di; if(hh<0||hh>=Hi) continue;
#pragma unroll
    for(int dj=0;dj<3;++dj){
      int ww=2*wo-1+dj; if(ww<0||ww>=Hi) continue;
      m=fmaxf(m,p[(long)hh*Hi+ww]);
    }
  }
  xout[idx]=m;
}

// ---------------- build xgf f16 [b][win][i][c], stride 264 ----------------
__global__ void k_mk_xgf(const float* __restrict__ xp2, f16* __restrict__ xgf){
  long total=8L*4*98*256;
  long idx=(long)blockIdx.x*256+threadIdx.x;
  if(idx>=total) return;
  int c=(int)(idx&255); long t=idx>>8;
  int i=(int)(t%98); t/=98;
  int win=(int)(t&3); int b=(int)(t>>2);
  int mi=i/49, ww=i%49, w1=ww/7, w2=ww%7;
  int nx=win>>1, ny=win&1;
  int X=nx*7+w1, Y=ny*7+w2;
  xgf[(((long)b*4+win)*98+i)*264+c]=(f16)xp2[(((long)(b*2+mi)*256+c)*14+X)*14+Y];
}

// ---------------- global attention: softmax(X X^T)*1/16 @ X, mean -> qg ----------------
__global__ __launch_bounds__(256) void k_gattn(const f16* __restrict__ xgf, float* __restrict__ qg){
  __shared__ __align__(16) f16 Xs[112][40];
  __shared__ __align__(16) f16 Pb[112][136];
  __shared__ __align__(16) f16 Bs[256][40];
  int bwn=blockIdx.x;                 // b*4+win
  int b=bwn>>2;
  const f16* Xg = xgf + (long)bwn*98*264;
  int tid=threadIdx.x, lane=tid&31, wv=tid>>5;
  for(int i=tid;i<112*136;i+=256) (&Pb[0][0])[i]=(f16)0.0f;

  v8f accS[7];
#pragma unroll
  for(int n=0;n<7;++n) accS[n]=v8f_zero();
  for(int k0=0;k0<256;k0+=32){
#pragma unroll
    for(int it=0;it<2;++it){
      int cidx=tid+it*256;
      if(cidx<448){
        int row=cidx>>2, ch=cidx&3;
        if(row<98) async_b128(lds_off(&Xs[row][ch*8]), Xg+(long)row*264+k0+ch*8);
        else       *(h8*)&Xs[row][ch*8]=h8_zero();
      }
    }
    wait_async0();
    __syncthreads();
    if(wv<7){
      v16h a=frag_a(&Xs[0][0],40,wv*16,0,lane);
#pragma unroll
      for(int n=0;n<7;++n){
        v16h bf=frag_b(&Xs[0][0],40,n*16,0,lane);
        accS[n]=WMMA16(a,bf,accS[n]);
      }
    }
    __syncthreads();
  }
  if(wv<7){
    int hi=lane>>4, nn=lane&15;
#pragma unroll
    for(int n=0;n<7;++n)
#pragma unroll
      for(int r=0;r<8;++r){
        int i_=wv*16+r+8*hi, j_=n*16+nn;
        if(i_<98 && j_<98) Pb[i_][j_]=(f16)accS[n][r];
      }
  }
  __syncthreads();
  // softmax rows, fold in C^-0.5 = 1/16
  if(tid<98){
    f16* rp=&Pb[tid][0];
    float mx=-1e30f;
    for(int j=0;j<98;++j) mx=fmaxf(mx,(float)rp[j]);
    float sm=0.0f;
    for(int j=0;j<98;++j) sm+=__expf((float)rp[j]-mx);
    float inv=0.0625f/sm;
    for(int j=0;j<98;++j) rp[j]=(f16)(__expf((float)rp[j]-mx)*inv);
  }
  __syncthreads();
  // O = P @ X  (K=98 padded to 128), accumulate mean into qg with atomics
  v8f accO[16];
#pragma unroll
  for(int n=0;n<16;++n) accO[n]=v8f_zero();
  for(int k0=0;k0<128;k0+=32){
    for(int i=tid;i<256*32;i+=256){
      int n=i>>5, kk=i&31;
      int j=k0+kk;
      Bs[n][kk]=(j<98)? Xg[(long)j*264+n] : (f16)0.0f;
    }
    __syncthreads();
    if(wv<7){
      v16h a=frag_a(&Pb[0][0],136,wv*16,k0,lane);
#pragma unroll
      for(int n=0;n<16;++n){
        v16h bf=frag_b(&Bs[0][0],40,n*16,0,lane);
        accO[n]=WMMA16(a,bf,accO[n]);
      }
    }
    __syncthreads();
  }
  if(wv<7){
    int hi=lane>>4, nn=lane&15;
#pragma unroll
    for(int n=0;n<16;++n)
#pragma unroll
      for(int r=0;r<8;++r){
        int i_=wv*16+r+8*hi;
        if(i_<98) atomicAdd(&qg[((long)b*98+i_)*256 + n*16+nn], accO[n][r]*0.25f);
      }
  }
}

// ---------------- KV projection: xw @ qkv_w + b -> k (natural), v (transposed) ----------------
__global__ __launch_bounds__(256) void k_kv(const float* __restrict__ x, const f16* __restrict__ qkvT,
                                            const float* __restrict__ qkvb,
                                            f16* __restrict__ kbuf, f16* __restrict__ vtbuf){
  __shared__ __align__(16) f16 As[112][40];
  __shared__ __align__(16) f16 Bs[64][40];
  int nblk=blockIdx.x, bw=blockIdx.y;
  int b=bw>>6, gxy=bw&63;
  int tid=threadIdx.x, lane=tid&31, wv=tid>>5;
  v8f acc[4];
#pragma unroll
  for(int n=0;n<4;++n) acc[n]=v8f_zero();
  for(int k0=0;k0<256;k0+=32){
    { int row=tid>>2, ch=tid&3;   // weight panel: async DMA to LDS
      async_b128(lds_off(&Bs[row][ch*8]), qkvT+(long)(nblk*64+row)*256+k0+ch*8); }
#pragma unroll
    for(int it=0;it<2;++it){      // activation panel: f32 -> f16 conversion
      int cidx=tid+it*256;
      if(cidx<448){
        int row=cidx>>2, ch=cidx&3;
        h8 v;
        if(row<98){
          long off=(((long)(b*2+row/49)*64+gxy)*49+(row%49))*256 + k0 + ch*8;
          const float* sp=x+off;
#pragma unroll
          for(int t=0;t<8;++t) v[t]=(f16)sp[t];
        } else v=h8_zero();
        *(h8*)&As[row][ch*8]=v;
      }
    }
    wait_async0();
    __syncthreads();
    if(wv<7){
      v16h a=frag_a(&As[0][0],40,wv*16,0,lane);
#pragma unroll
      for(int n=0;n<4;++n){
        v16h bf=frag_b(&Bs[0][0],40,n*16,0,lane);
        acc[n]=WMMA16(a,bf,acc[n]);
      }
    }
    __syncthreads();
  }
  if(wv<7){
    int hi=lane>>4, nn=lane&15;
#pragma unroll
    for(int n=0;n<4;++n)
#pragma unroll
      for(int r=0;r<8;++r){
        int j=wv*16+r+8*hi;
        int col=nblk*64+n*16+nn;
        if(j<98){
          float val=acc[n][r]+qkvb[col];
          int h=(col&255)>>5, e=col&31;
          if(col<256) kbuf[(((long)bw*8+h)*98+j)*32+e]=(f16)val;
          else        vtbuf[(((long)bw*8+h)*32+e)*104+j]=(f16)val;
        }
      }
  }
}

// ---------------- window attention per (head, window) ----------------
__global__ __launch_bounds__(256) void k_attn(const float* __restrict__ qg, const f16* __restrict__ kbuf,
                                              const f16* __restrict__ vtbuf, const float* __restrict__ rpe,
                                              f16* __restrict__ aobuf){
  __shared__ __align__(16) f16 Qs[112][40];
  __shared__ __align__(16) f16 Ks[112][40];
  __shared__ __align__(16) f16 Pb[112][136];
  __shared__ __align__(16) f16 Vs[32][136];
  int h=blockIdx.x, bw=blockIdx.y;
  int b=bw>>6;
  int tid=threadIdx.x, lane=tid&31, wv=tid>>5;
  const float scale=0.17677669529663687f;   // 32^-0.5
  for(int i=tid;i<112*40;i+=256){
    int row=i/40, cc=i%40;
    f16 qv=(f16)0.0f, kv=(f16)0.0f;
    if(row<98 && cc<32){
      qv=(f16)(qg[((long)b*98+row)*256 + h*32+cc]*scale);
      kv=kbuf[(((long)bw*8+h)*98+row)*32+cc];
    }
    Qs[row][cc]=qv; Ks[row][cc]=kv;
  }
  for(int i=tid;i<32*136;i+=256){
    int e=i/136, j=i%136;
    Vs[e][j]=(j<98)? vtbuf[(((long)bw*8+h)*32+e)*104+j] : (f16)0.0f;
  }
  for(int i=tid;i<112*136;i+=256) (&Pb[0][0])[i]=(f16)0.0f;
  __syncthreads();
  // S = (q*scale) k^T + rpe bias
  if(wv<7){
    v16h a=frag_a(&Qs[0][0],40,wv*16,0,lane);
    int hi=lane>>4, nn=lane&15;
#pragma unroll
    for(int n=0;n<7;++n){
      v16h bf=frag_b(&Ks[0][0],40,n*16,0,lane);
      v8f acc=v8f_zero();
      acc=WMMA16(a,bf,acc);
#pragma unroll
      for(int r=0;r<8;++r){
        int i_=wv*16+r+8*hi, j_=n*16+nn;
        if(i_<98 && j_<98){
          int mi=i_/49, r1=(i_%49)/7, c1=i_%7;
          int mj=j_/49, r2=(j_%49)/7, c2=j_%7;
          int idx=(mi-mj+1)*169 + (r1-r2+6)*13 + (c1-c2+6);
          Pb[i_][j_]=(f16)(acc[r]+rpe[idx*8+h]);
        }
      }
    }
  }
  __syncthreads();
  if(tid<98){
    f16* rp=&Pb[tid][0];
    float mx=-1e30f;
    for(int j=0;j<98;++j) mx=fmaxf(mx,(float)rp[j]);
    float sm=0.0f;
    for(int j=0;j<98;++j) sm+=__expf((float)rp[j]-mx);
    float inv=1.0f/sm;
    for(int j=0;j<98;++j) rp[j]=(f16)(__expf((float)rp[j]-mx)*inv);
  }
  __syncthreads();
  // O = P @ V   (K = 98 padded to 128)
  if(wv<7){
    v8f o0=v8f_zero(), o1=v8f_zero();
#pragma unroll
    for(int k0=0;k0<128;k0+=32){
      v16h a=frag_a(&Pb[0][0],136,wv*16,k0,lane);
      v16h b0=frag_b(&Vs[0][0],136,0,k0,lane);
      v16h b1=frag_b(&Vs[0][0],136,16,k0,lane);
      o0=WMMA16(a,b0,o0); o1=WMMA16(a,b1,o1);
    }
    int hi=lane>>4, nn=lane&15;
#pragma unroll
    for(int r=0;r<8;++r){
      int i_=wv*16+r+8*hi;
      if(i_<98){
        long rb=((long)bw*98+i_)*256 + h*32;
        aobuf[rb+nn]=(f16)o0[r];
        aobuf[rb+16+nn]=(f16)o1[r];
      }
    }
  }
}

// ---------------- output projection GEMM + scatter to final layout ----------------
__global__ __launch_bounds__(256) void k_toout(const f16* __restrict__ aobuf, const f16* __restrict__ toT,
                                               float* __restrict__ out){
  __shared__ __align__(16) f16 As[64][40];
  __shared__ __align__(16) f16 Bs[64][40];
  int mbase=blockIdx.x*64, nbase=blockIdx.y*64;
  int tid=threadIdx.x, lane=tid&31, wv=tid>>5;
  int wm=wv>>1, wn=wv&1;
  v8f acc0=v8f_zero(), acc1=v8f_zero();
  for(int k0=0;k0<256;k0+=32){
    int row=tid>>2, ch=tid&3;
    async_b128(lds_off(&As[row][ch*8]), aobuf+(long)(mbase+row)*256+k0+ch*8);
    async_b128(lds_off(&Bs[row][ch*8]), toT  +(long)(nbase+row)*256+k0+ch*8);
    wait_async0();
    __syncthreads();
    v16h a =frag_a(&As[0][0],40,wm*16,0,lane);
    v16h b0=frag_b(&Bs[0][0],40,wn*32,0,lane);
    v16h b1=frag_b(&Bs[0][0],40,wn*32+16,0,lane);
    acc0=WMMA16(a,b0,acc0); acc1=WMMA16(a,b1,acc1);
    __syncthreads();
  }
  int hi=lane>>4, nn=lane&15;
#pragma unroll
  for(int r=0;r<8;++r){
    int rg=mbase+wm*16+r+8*hi;       // global row: bw*98 + i
    int bw=rg/98, i=rg%98;
    int b=bw>>6, gxy=bw&63;
    int mi=i/49, ww=i%49;
    int c0=nbase+wn*32+nn;
    long off=(((long)(b*2+mi)*64+gxy)*49+ww)*256 + c0;
    out[off]=acc0[r];
    out[off+16]=acc1[r];
  }
}

// ---------------- launch ----------------
extern "C" void kernel_launch(void* const* d_in, const int* in_sizes, int n_in,
                              void* d_out, int out_size, void* d_ws, size_t ws_size,
                              hipStream_t stream){
  const float* x    = (const float*)d_in[0];
  const float* qkvw = (const float*)d_in[1];
  const float* qkvb = (const float*)d_in[2];
  const float* tow  = (const float*)d_in[3];
  const float* rpe  = (const float*)d_in[4];
  const float* dw1  = (const float*)d_in[5];
  const float* se11 = (const float*)d_in[6];
  const float* se12 = (const float*)d_in[7];
  const float* pw1  = (const float*)d_in[8];
  const float* dw2  = (const float*)d_in[9];
  const float* se21 = (const float*)d_in[10];
  const float* se22 = (const float*)d_in[11];
  const float* pw2  = (const float*)d_in[12];
  float* out = (float*)d_out;

  char* ws=(char*)d_ws;
  size_t o=0;
  auto take=[&](size_t bytes)->size_t{ size_t r=o; o=(o+bytes+255)&~(size_t)255; return r; };
  size_t off_xg0 = take(16L*256*3136*4);     // stage1 feature map (residual in-place); later aliased by kbuf
  size_t off_yb  = take(16L*256*3136*4);     // conv+gelu output; later aliased by vtbuf
  size_t off_yt  = take(16L*3136*256*2);     // f16 transposed scaled; later aliased by aobuf
  size_t off_xp1 = take(16L*256*784*4);
  size_t off_xp2 = take(16L*256*196*4);
  size_t off_xgf = take(32L*98*264*2);
  size_t off_qg  = take(8L*98*256*4);
  size_t off_qkT = take(512L*256*2);
  size_t off_toT = take(256L*256*2);
  size_t off_pw1 = take(256L*256*2);
  size_t off_pw2 = take(256L*256*2);
  size_t off_s0  = take(16L*256*4);
  size_t off_sb  = take(16L*256*4);
  (void)ws_size; (void)n_in; (void)in_sizes; (void)out_size;

  float* xg0=(float*)(ws+off_xg0);
  float* yb =(float*)(ws+off_yb);
  f16*   yt =(f16*)(ws+off_yt);
  float* xp1=(float*)(ws+off_xp1);
  float* xp2=(float*)(ws+off_xp2);
  f16*   xgf=(f16*)(ws+off_xgf);
  float* qg =(float*)(ws+off_qg);
  f16*   qkvT=(f16*)(ws+off_qkT);
  f16*   toT=(f16*)(ws+off_toT);
  f16*   pw1h=(f16*)(ws+off_pw1);
  f16*   pw2h=(f16*)(ws+off_pw2);
  float* s0 =(float*)(ws+off_s0);
  float* sb =(float*)(ws+off_sb);
  // aliases (stage-1 buffers are dead by the time these are written)
  f16* kbuf=(f16*)(ws+off_xg0);   // 512*8*98*32 halves
  f16* vtbuf=(f16*)(ws+off_yb);   // 512*8*32*104 halves
  f16* aobuf=(f16*)(ws+off_yt);   // 512*98*256 halves

  // 0) weight prep + qg zero
  k_prep<<<784,256,0,stream>>>(qkvw,tow,pw1,pw2,qkvT,toT,pw1h,pw2h,qg);
  // 1) rearrange input
  k_rearrange_in<<<50176,256,0,stream>>>(x,xg0);
  // 2) stage 1 feat_extract @56
  k_dwconv_gelu<<<50176,256,0,stream>>>(xg0,dw1,yb,56);
  k_chan_mean<<<4096,256,0,stream>>>(yb,s0,3136);
  k_se<<<16,256,0,stream>>>(s0,se11,se12,sb);
  k_yscale_t<<<50176,256,0,stream>>>(yb,sb,yt,3136);
  { dim3 g(49,4,16); k_pwconv<<<g,256,0,stream>>>(yt,pw1h,xg0,3136); }
  k_maxpool<<<12544,256,0,stream>>>(xg0,xp1,56);
  // 3) stage 2 feat_extract @28
  k_dwconv_gelu<<<12544,256,0,stream>>>(xp1,dw2,yb,28);
  k_chan_mean<<<4096,256,0,stream>>>(yb,s0,784);
  k_se<<<16,256,0,stream>>>(s0,se21,se22,sb);
  k_yscale_t<<<12544,256,0,stream>>>(yb,sb,yt,784);
  { dim3 g(13,4,16); k_pwconv<<<g,256,0,stream>>>(yt,pw2h,xp1,784); }
  k_maxpool<<<3136,256,0,stream>>>(xp1,xp2,28);
  // 4) global query path
  k_mk_xgf<<<3136,256,0,stream>>>(xp2,xgf);
  k_gattn<<<32,256,0,stream>>>(xgf,qg);
  // 5) window attention
  { dim3 g(8,512); k_kv<<<g,256,0,stream>>>(x,qkvT,qkvb,kbuf,vtbuf); }
  { dim3 g(8,512); k_attn<<<g,256,0,stream>>>(qg,kbuf,vtbuf,rpe,aobuf); }
  // 6) output projection + scatter
  { dim3 g(784,4); k_toout<<<g,256,0,stream>>>(aobuf,toT,out); }
}
